// CausalSelfAttention_22660247453912
// MI455X (gfx1250) — matrix operand order repaired and checked by reading, not compiled, not run on hardware
//
#include <hip/hip_runtime.h>
#include <hip/hip_bf16.h>

typedef __attribute__((ext_vector_type(4)))  _Float16 v4h;
typedef __attribute__((ext_vector_type(8)))  _Float16 v8h;
typedef __attribute__((ext_vector_type(16))) _Float16 v16h;
typedef __attribute__((ext_vector_type(8)))  float    v8f;
typedef __attribute__((ext_vector_type(4)))  int      v4i;

#define BB   4
#define TT   2048
#define CC   1024
#define HH   16
#define HD   64
#define MTOT (BB * TT)   // 8192 rows of x / y

union U8F { v8f v; float f[8]; };
union U8H { v8h v; _Float16 h[8]; };

// D = A*B + C, f16 inputs, f32 accum
#define WMMA(a, b, c) __builtin_amdgcn_wmma_f32_16x16x32_f16(false, (a), false, (b), (short)0, (c), false, false)

// ---- gfx1250 async global->LDS path (ASYNCcnt), compile-safe fallback ------
#if defined(__gfx1250__) && __has_builtin(__builtin_amdgcn_global_load_async_to_lds_b128)
#define HAS_ASYNC_LDS 1
__device__ __forceinline__ void async_copy16(const void* g, void* l) {
    // builtin signature (probe-confirmed by diagnostic): (v4i AS1*, v4i AS3*, imm, imm)
    __builtin_amdgcn_global_load_async_to_lds_b128(
        (__attribute__((address_space(1))) v4i*)g,
        (__attribute__((address_space(3))) v4i*)l, 0, 0);
}
__device__ __forceinline__ void async_wait0() {
    asm volatile("s_wait_asynccnt 0x0" ::: "memory");
}
#else
#define HAS_ASYNC_LDS 0
__device__ __forceinline__ void async_copy16(const void* g, void* l) {
    *(v8h*)l = *(const v8h*)g;          // 16B synchronous copy fallback
}
__device__ __forceinline__ void async_wait0() {}
#endif

// A-matrix fragment (16x32 f16): lane holds row m = lane&15.
// halves 0..7  = k in [kk+8*hi, +7],  halves 8..15 = k in [kk+16+8*hi, +7]
__device__ __forceinline__ v16h frag_a(const _Float16* row, int kk, int hi) {
    v16h r;
    ((v8h*)&r)[0] = *(const v8h*)(row + kk + 8 * hi);
    ((v8h*)&r)[1] = *(const v8h*)(row + kk + 16 + 8 * hi);
    return r;
}
// B-matrix fragment (32x16 f16), B[k][n] = src_row_n[k]: lane holds col n = lane&15,
// halves 0..7 = k in [kk+8*hi, +7], halves 8..15 = k in [kk+16+8*hi, +7]
__device__ __forceinline__ v16h frag_b(const _Float16* row, int kk, int hi) {
    v16h r;
    ((v8h*)&r)[0] = *(const v8h*)(row + kk + 8 * hi);
    ((v8h*)&r)[1] = *(const v8h*)(row + kk + 16 + 8 * hi);
    return r;
}

// ---------------------------------------------------------------------------
// Kernel 1: QKV projection. out[m,n] = sum_k x[m,k] * W[n,k] + b[n]
// BM=128, BN=128, BK=64; 8 waves, each 32x64. blockIdx.z: 0=Q(*1/8), 1=K, 2=V.
// Output f16 in [B,H,T,HD].
// ---------------------------------------------------------------------------
__global__ __launch_bounds__(256) void qkv_gemm(
    const float* __restrict__ x,
    const float* __restrict__ Wq, const float* __restrict__ bq,
    const float* __restrict__ Wk, const float* __restrict__ bk,
    const float* __restrict__ Wv, const float* __restrict__ bv,
    _Float16* __restrict__ qb, _Float16* __restrict__ kb, _Float16* __restrict__ vb)
{
    const int z = blockIdx.z;
    const float* W    = (z == 0) ? Wq : (z == 1) ? Wk : Wv;
    const float* bias = (z == 0) ? bq : (z == 1) ? bk : bv;
    _Float16*    dst  = (z == 0) ? qb : (z == 1) ? kb : vb;
    const float scale = (z == 0) ? 0.125f : 1.0f;   // 1/sqrt(64)

    __shared__ _Float16 Al[128][72];
    __shared__ _Float16 Bl[128][72];

    const int tid  = threadIdx.x;
    const int lane = tid & 31, wid = tid >> 5;
    const int hi   = lane >> 4, ln = lane & 15;
    const int wm   = wid >> 1,  wn = wid & 1;      // 4x2 wave grid, 32x64 per wave
    const int m0   = blockIdx.x * 128, n0 = blockIdx.y * 128;

    v8f acc[2][4] = {};

    for (int kk = 0; kk < CC; kk += 64) {
#pragma unroll
        for (int it = 0; it < 8; ++it) {           // A: 128x64 f32 -> f16
            int idx = it * 256 + tid;
            int r = idx >> 4, c = (idx & 15) * 4;
            float4 f = *(const float4*)(x + (size_t)(m0 + r) * CC + kk + c);
            v4h h4 = { (_Float16)f.x, (_Float16)f.y, (_Float16)f.z, (_Float16)f.w };
            *(v4h*)&Al[r][c] = h4;
        }
#pragma unroll
        for (int it = 0; it < 8; ++it) {           // W: 128x64 f32 -> f16
            int idx = it * 256 + tid;
            int r = idx >> 4, c = (idx & 15) * 4;
            float4 f = *(const float4*)(W + (size_t)(n0 + r) * CC + kk + c);
            v4h h4 = { (_Float16)f.x, (_Float16)f.y, (_Float16)f.z, (_Float16)f.w };
            *(v4h*)&Bl[r][c] = h4;
        }
        __syncthreads();

#pragma unroll
        for (int ks = 0; ks < 2; ++ks) {
            v16h af[2], bf[4];
#pragma unroll
            for (int mt = 0; mt < 2; ++mt)
                af[mt] = frag_a(&Al[wm * 32 + mt * 16 + ln][0], ks * 32, hi);
#pragma unroll
            for (int nt = 0; nt < 4; ++nt)
                bf[nt] = frag_b(&Bl[wn * 64 + nt * 16 + ln][0], ks * 32, hi);
#pragma unroll
            for (int mt = 0; mt < 2; ++mt)
#pragma unroll
                for (int nt = 0; nt < 4; ++nt)
                    acc[mt][nt] = WMMA(af[mt], bf[nt], acc[mt][nt]);
        }
        __syncthreads();
    }

#pragma unroll
    for (int mt = 0; mt < 2; ++mt)
#pragma unroll
        for (int nt = 0; nt < 4; ++nt) {
            U8F u; u.v = acc[mt][nt];
            int n = n0 + wn * 64 + nt * 16 + ln;
            float bval = bias[n];
            int h = n >> 6, d = n & 63;
#pragma unroll
            for (int r = 0; r < 8; ++r) {
                int m = m0 + wm * 32 + mt * 16 + r + 8 * hi;
                int b = m >> 11, t = m & (TT - 1);
                dst[(((size_t)(b * HH + h)) * TT + t) * HD + d] =
                    (_Float16)((u.f[r] + bval) * scale);
            }
        }
}

// ---------------------------------------------------------------------------
// Kernel 2: flash attention. One block = 64 q-rows of one (b,h). 4 waves,
// each wave owns 16 q-rows. Online softmax in f32, S and PV via WMMA f16.
// Q/K tiles use the async global->LDS path (f16, no conversion needed).
// ---------------------------------------------------------------------------
__global__ __launch_bounds__(128) void attn_fa(
    const _Float16* __restrict__ q, const _Float16* __restrict__ k,
    const _Float16* __restrict__ v, _Float16* __restrict__ y)
{
    __shared__ _Float16 Ql[64][72];
    __shared__ _Float16 Kl[64][72];
    __shared__ _Float16 VlT[64][72];        // transposed: VlT[d][key]
    __shared__ _Float16 Pl[4][16][72];      // wave-private P tiles

    const int tid  = threadIdx.x;
    const int lane = tid & 31, wid = tid >> 5;
    const int hi   = lane >> 4, ln = lane & 15;
    const int qblk = blockIdx.x;
    const int bh   = blockIdx.y;
    const size_t base = (size_t)bh * TT * HD;

#pragma unroll
    for (int it = 0; it < 4; ++it) {        // Q block 64x64, async -> LDS
        int idx = it * 128 + tid;
        int r = idx >> 3, c = (idx & 7) * 8;
        async_copy16(q + base + (size_t)(qblk * 64 + r) * HD + c, &Ql[r][c]);
    }
    async_wait0();
    __syncthreads();

    v16h aq[2];
    aq[0] = frag_a(&Ql[wid * 16 + ln][0], 0, hi);
    aq[1] = frag_a(&Ql[wid * 16 + ln][0], 32, hi);

    float rmax[8], rsum[8];
#pragma unroll
    for (int r = 0; r < 8; ++r) { rmax[r] = -1e30f; rsum[r] = 0.0f; }
    v8f oacc[4] = {};

    for (int j = 0; j <= qblk; ++j) {
#pragma unroll
        for (int it = 0; it < 4; ++it) {    // K block 64x64 (row-major), async
            int idx = it * 128 + tid;
            int r = idx >> 3, c = (idx & 7) * 8;
            async_copy16(k + base + (size_t)(j * 64 + r) * HD + c, &Kl[r][c]);
        }
#pragma unroll
        for (int it = 0; it < 4; ++it) {    // V block, transposed into VlT[d][key]
            int idx = it * 128 + tid;
            int r = idx >> 3, c = (idx & 7) * 8;
            U8H ch; ch.v = *(const v8h*)(v + base + (size_t)(j * 64 + r) * HD + c);
#pragma unroll
            for (int e = 0; e < 8; ++e) VlT[c + e][r] = ch.h[e];
        }
        async_wait0();
        __syncthreads();

        // S = Q * K^T  (16x64 per wave)
        v8f s[4] = {};
#pragma unroll
        for (int nt = 0; nt < 4; ++nt) {
            v16h b0 = frag_b(&Kl[nt * 16 + ln][0], 0, hi);
            v16h b1 = frag_b(&Kl[nt * 16 + ln][0], 32, hi);
            s[nt] = WMMA(aq[0], b0, s[nt]);
            s[nt] = WMMA(aq[1], b1, s[nt]);
        }

        U8F su[4];
#pragma unroll
        for (int nt = 0; nt < 4; ++nt) su[nt].v = s[nt];

        if (j == qblk) {                    // causal mask on diagonal block
#pragma unroll
            for (int nt = 0; nt < 4; ++nt) {
                int ki = j * 64 + nt * 16 + ln;
#pragma unroll
                for (int r = 0; r < 8; ++r) {
                    int qi = qblk * 64 + wid * 16 + r + 8 * hi;
                    if (ki > qi) su[nt].f[r] = -1e30f;
                }
            }
        }

        // online softmax: row r+8*hi spans lane&15 across a 16-lane half
        float corr[8];
#pragma unroll
        for (int r = 0; r < 8; ++r) {
            float mx = fmaxf(fmaxf(su[0].f[r], su[1].f[r]), fmaxf(su[2].f[r], su[3].f[r]));
            mx = fmaxf(mx, __shfl_xor(mx, 1, 32));
            mx = fmaxf(mx, __shfl_xor(mx, 2, 32));
            mx = fmaxf(mx, __shfl_xor(mx, 4, 32));
            mx = fmaxf(mx, __shfl_xor(mx, 8, 32));
            float mnew = fmaxf(rmax[r], mx);
            corr[r] = __expf(rmax[r] - mnew);
            rmax[r] = mnew;
            float ps = 0.0f;
#pragma unroll
            for (int nt = 0; nt < 4; ++nt) {
                float p = __expf(su[nt].f[r] - mnew);
                su[nt].f[r] = p;
                ps += p;
            }
            ps += __shfl_xor(ps, 1, 32);
            ps += __shfl_xor(ps, 2, 32);
            ps += __shfl_xor(ps, 4, 32);
            ps += __shfl_xor(ps, 8, 32);
            rsum[r] = rsum[r] * corr[r] + ps;
        }
#pragma unroll
        for (int nt = 0; nt < 4; ++nt) {
            U8F ou; ou.v = oacc[nt];
#pragma unroll
            for (int r = 0; r < 8; ++r) ou.f[r] *= corr[r];
            oacc[nt] = ou.v;
        }

        // re-layout P (C/D layout -> A layout) through wave-private LDS
#pragma unroll
        for (int nt = 0; nt < 4; ++nt)
#pragma unroll
            for (int r = 0; r < 8; ++r)
                Pl[wid][r + 8 * hi][nt * 16 + ln] = (_Float16)su[nt].f[r];
        asm volatile("s_wait_dscnt 0x0" ::: "memory");   // cross-lane store->load, same wave

        // O += P * V
#pragma unroll
        for (int ks = 0; ks < 2; ++ks) {
            v16h ap = frag_a(&Pl[wid][ln][0], ks * 32, hi);
#pragma unroll
            for (int nt = 0; nt < 4; ++nt) {
                v16h bv = frag_b(&VlT[nt * 16 + ln][0], ks * 32, hi);
                oacc[nt] = WMMA(ap, bv, oacc[nt]);
            }
        }
        __syncthreads();                    // protect Kl/VlT before next j
    }

    // normalize and store y in [B,T,C] (f16) for the projection GEMM
    const int b = bh >> 4, h = bh & 15;
#pragma unroll
    for (int nt = 0; nt < 4; ++nt) {
        U8F ou; ou.v = oacc[nt];
#pragma unroll
        for (int r = 0; r < 8; ++r) {
            int t = qblk * 64 + wid * 16 + r + 8 * hi;
            int d = nt * 16 + ln;
            y[((size_t)(b * TT + t)) * CC + h * HD + d] = (_Float16)(ou.f[r] / rsum[r]);
        }
    }
}

// ---------------------------------------------------------------------------
// Kernel 3: output projection. out[m,n] = sum_k y16[m,k] * Wp[n,k] + bp[n], f32 out
// Same 128x128x64 tiling; A tile (f16) uses async global->LDS.
// ---------------------------------------------------------------------------
__global__ __launch_bounds__(256) void proj_gemm(
    const _Float16* __restrict__ yin, const float* __restrict__ Wp,
    const float* __restrict__ bp, float* __restrict__ out)
{
    __shared__ _Float16 Al[128][72];
    __shared__ _Float16 Bl[128][72];

    const int tid  = threadIdx.x;
    const int lane = tid & 31, wid = tid >> 5;
    const int hi   = lane >> 4, ln = lane & 15;
    const int wm   = wid >> 1,  wn = wid & 1;
    const int m0   = blockIdx.x * 128, n0 = blockIdx.y * 128;

    v8f acc[2][4] = {};

    for (int kk = 0; kk < CC; kk += 64) {
#pragma unroll
        for (int it = 0; it < 4; ++it) {    // A: 128x64 f16, async -> LDS
            int idx = it * 256 + tid;       // 1024 16B chunks
            int r = idx >> 3, c = (idx & 7) * 8;
            async_copy16(yin + (size_t)(m0 + r) * CC + kk + c, &Al[r][c]);
        }
#pragma unroll
        for (int it = 0; it < 8; ++it) {    // W: 128x64 f32 -> f16
            int idx = it * 256 + tid;
            int r = idx >> 4, c = (idx & 15) * 4;
            float4 f = *(const float4*)(Wp + (size_t)(n0 + r) * CC + kk + c);
            v4h h4 = { (_Float16)f.x, (_Float16)f.y, (_Float16)f.z, (_Float16)f.w };
            *(v4h*)&Bl[r][c] = h4;
        }
        async_wait0();
        __syncthreads();

#pragma unroll
        for (int ks = 0; ks < 2; ++ks) {
            v16h af[2], bf[4];
#pragma unroll
            for (int mt = 0; mt < 2; ++mt)
                af[mt] = frag_a(&Al[wm * 32 + mt * 16 + ln][0], ks * 32, hi);
#pragma unroll
            for (int nt = 0; nt < 4; ++nt)
                bf[nt] = frag_b(&Bl[wn * 64 + nt * 16 + ln][0], ks * 32, hi);
#pragma unroll
            for (int mt = 0; mt < 2; ++mt)
#pragma unroll
                for (int nt = 0; nt < 4; ++nt)
                    acc[mt][nt] = WMMA(af[mt], bf[nt], acc[mt][nt]);
        }
        __syncthreads();
    }

#pragma unroll
    for (int mt = 0; mt < 2; ++mt)
#pragma unroll
        for (int nt = 0; nt < 4; ++nt) {
            U8F u; u.v = acc[mt][nt];
            int n = n0 + wn * 64 + nt * 16 + ln;
            float bval = bp[n];
#pragma unroll
            for (int r = 0; r < 8; ++r) {
                int m = m0 + wm * 32 + mt * 16 + r + 8 * hi;
                out[(size_t)m * CC + n] = u.f[r] + bval;
            }
        }
}

// ---------------------------------------------------------------------------
extern "C" void kernel_launch(void* const* d_in, const int* in_sizes, int n_in,
                              void* d_out, int out_size, void* d_ws, size_t ws_size,
                              hipStream_t stream) {
    (void)in_sizes; (void)n_in; (void)out_size; (void)ws_size;
    const float* x  = (const float*)d_in[0];
    const float* Wk = (const float*)d_in[1];
    const float* bk = (const float*)d_in[2];
    const float* Wq = (const float*)d_in[3];
    const float* bq = (const float*)d_in[4];
    const float* Wv = (const float*)d_in[5];
    const float* bv = (const float*)d_in[6];
    const float* Wp = (const float*)d_in[7];
    const float* bp = (const float*)d_in[8];
    float* out = (float*)d_out;

    _Float16* qb = (_Float16*)d_ws;                 // [B,H,T,HD] f16, pre-scaled
    _Float16* kb = qb + (size_t)MTOT * CC;          // [B,H,T,HD] f16
    _Float16* vb = kb + (size_t)MTOT * CC;          // [B,H,T,HD] f16
    _Float16* yb = vb + (size_t)MTOT * CC;          // [B,T,C]   f16

    dim3 g1(MTOT / 128, CC / 128, 3);
    qkv_gemm<<<g1, 256, 0, stream>>>(x, Wq, bq, Wk, bk, Wv, bv, qb, kb, vb);

    dim3 g2(TT / 64, BB * HH);
    attn_fa<<<g2, 128, 0, stream>>>(qb, kb, vb, yb);

    dim3 g3(MTOT / 128, CC / 128);
    proj_gemm<<<g3, 256, 0, stream>>>(yb, Wp, bp, out);
}
